// Encoder_44925357916237
// MI455X (gfx1250) — compile-verified, matrix operand order checked by reading, not run on hardware
//
#include <hip/hip_runtime.h>
#include <hip/hip_bf16.h>
#include <stdint.h>
#include <math.h>

typedef _Float16 f16;
typedef __attribute__((ext_vector_type(16))) _Float16 v16h;
typedef __attribute__((ext_vector_type(8)))  float    v8f;
typedef __attribute__((ext_vector_type(4)))  unsigned int tdm_g0_t;
typedef __attribute__((ext_vector_type(8)))  int          tdm_g1_t;
typedef __attribute__((ext_vector_type(4)))  int          tdm_g2_t;

#define B_   32
#define N_   64
#define D_   256
#define H_   256
#define BN_  (B_ * N_)     // 2048
#define G3_  (3 * H_)      // 768

__device__ __forceinline__ float sigmoidf_(float x) {
    return 1.0f / (1.0f + __expf(-x));
}

// ---------------------------------------------------------------------------
// Per-lane fragment (A or B operand) for v_wmma_f32_16x16x32_f16 from a
// row-major f16 matrix (row = M-row for A, N-col for B^T).  ISA 7.12.2:
//   lanes 0-15 : row = lane,    K-halves [0..8)  and [16..24)
//   lanes 16-31: row = lane-16, K-halves [8..16) and [24..32)
// ---------------------------------------------------------------------------
__device__ __forceinline__ v16h load_frag16(const f16* base, int row, int ld,
                                            int k0, int lane) {
    const int kb = (lane >> 4) * 8;
    const f16* p = base + (size_t)row * ld + k0;
    v16h f;
#pragma unroll
    for (int j = 0; j < 8; ++j) {
        const int kk = (j < 4) ? (kb + 2 * j) : (16 + kb + 2 * (j - 4));
        union { uint32_t u; f16 h[2]; } cv;
        cv.u = *reinterpret_cast<const uint32_t*>(p + kk);
        f[2 * j]     = cv.h[0];
        f[2 * j + 1] = cv.h[1];
    }
    return f;
}

// ---------------------------------------------------------------------------
// Weight convert: out_h[n*K + k] = (f16) in[k*N + n]   (transpose K x N -> N x K)
// ---------------------------------------------------------------------------
__global__ void k_wtrans(const float* __restrict__ in, f16* __restrict__ out,
                         int K, int N) {
    int i = blockIdx.x * blockDim.x + threadIdx.x;
    if (i >= K * N) return;
    int n = i / K;
    int k = i - n * K;
    out[i] = (f16)in[(size_t)k * N + n];
}

// ---------------------------------------------------------------------------
// Embedding gather: emb_h[bn, 0:256] = emb_W[xs[bn]], [256:512] = emb_W[labels[bn]]
// ---------------------------------------------------------------------------
__global__ void k_gather(const int* __restrict__ xs, const int* __restrict__ labels,
                         const float* __restrict__ embW, f16* __restrict__ embH) {
    int i = blockIdx.x * blockDim.x + threadIdx.x;
    if (i >= BN_ * 2 * D_) return;
    int bn = i >> 9;
    int d  = i & 511;
    int idx = (d < D_) ? xs[bn] : labels[bn];
    embH[i] = (f16)embW[(size_t)idx * D_ + (d & (D_ - 1))];
}

// ---------------------------------------------------------------------------
// outputs_h[bn, 0:256] = grus[bn], outputs_h[bn, 256:512] = rel_W[rels[bn]]
// ---------------------------------------------------------------------------
__global__ void k_build_outputs(const float* __restrict__ grus, const int* __restrict__ rels,
                                const float* __restrict__ relW, f16* __restrict__ outH) {
    int i = blockIdx.x * blockDim.x + threadIdx.x;
    if (i >= BN_ * 2 * D_) return;
    int bn = i >> 9;
    int d  = i & 511;
    float v = (d < D_) ? grus[(size_t)bn * D_ + d]
                       : relW[(size_t)rels[bn] * D_ + (d - D_)];
    outH[i] = (f16)v;
}

// ---------------------------------------------------------------------------
// WMMA GEMM: C[M,N] = A_h[M,K] * Bt_h[N,K]^T + bias[N]   (f16 in, f32 out)
// One wave computes a 16x64 strip (4 N-tiles sharing one A fragment);
// block = 8 waves covering 32(M) x 256(N).
// ---------------------------------------------------------------------------
__global__ __launch_bounds__(256)
void k_gemm_bias(const f16* __restrict__ A, const f16* __restrict__ Bt,
                 const float* __restrict__ bias, float* __restrict__ C,
                 int M, int N, int K) {
    const int lane = threadIdx.x & 31;
    const int wave = threadIdx.x >> 5;                 // 0..7
    const int m0 = blockIdx.y * 32 + (wave >> 2) * 16;
    const int n0 = blockIdx.x * 256 + (wave & 3) * 64;
    if (m0 >= M || n0 >= N) return;

    v8f c[4];
#pragma unroll
    for (int t = 0; t < 4; ++t) c[t] = (v8f){};

    const int ar = m0 + (lane & 15);
    const int br = n0 + (lane & 15);
    for (int k0 = 0; k0 < K; k0 += 32) {
        // prefetch next K-slab (speculative; OOB prefetch is dropped)
        __builtin_prefetch(A  + (size_t)ar * K + k0 + 32, 0, 1);
        __builtin_prefetch(Bt + (size_t)br * K + k0 + 32, 0, 1);
        v16h a = load_frag16(A, ar, K, k0, lane);
#pragma unroll
        for (int t = 0; t < 4; ++t) {
            v16h b = load_frag16(Bt, br + t * 16, K, k0, lane);
            c[t] = __builtin_amdgcn_wmma_f32_16x16x32_f16(false, a, false, b,
                                                          (short)0, c[t], false, false);
        }
    }
#pragma unroll
    for (int t = 0; t < 4; ++t) {
        const int col = n0 + t * 16 + (lane & 15);
        const float bv = bias[col];
#pragma unroll
        for (int v = 0; v < 8; ++v) {
            const int row = m0 + 8 * (lane >> 4) + v;
            C[(size_t)row * N + col] = c[t][v] + bv;
        }
    }
}

// ---------------------------------------------------------------------------
// Shared recurrent-GEMM helper: wave owns M-tile m0 and N-tiles ntA, ntA+1;
// computes 6 accumulators (2 tiles x 3 gates) sharing one A fragment per K-step.
// ---------------------------------------------------------------------------
__device__ __forceinline__ void rec_gemm6(const f16* h16f, const f16* WT,
                                          int m0, int ntA, int lane,
                                          v8f c[2][3]) {
    const int ar = m0 + (lane & 15);
#pragma unroll
    for (int tr = 0; tr < 2; ++tr)
#pragma unroll
        for (int g = 0; g < 3; ++g) c[tr][g] = (v8f){};

    for (int k0 = 0; k0 < H_; k0 += 32) {
        v16h a = load_frag16(h16f, ar, H_, k0, lane);
#pragma unroll
        for (int tr = 0; tr < 2; ++tr) {
            const int nc = (ntA + tr) * 16 + (lane & 15);
#pragma unroll
            for (int g = 0; g < 3; ++g) {
                v16h b = load_frag16(WT, g * H_ + nc, H_, k0, lane);
                c[tr][g] = __builtin_amdgcn_wmma_f32_16x16x32_f16(false, a, false, b,
                                                                  (short)0, c[tr][g],
                                                                  false, false);
            }
        }
    }
}

// ---------------------------------------------------------------------------
// Persistent bidirectional GRU scan.  grid.x = 2 (0: forward, 1: backward).
// ---------------------------------------------------------------------------
__global__ __launch_bounds__(512)
void k_bigru_scan(const float* __restrict__ xp_f, const float* __restrict__ xp_b,
                  const f16* __restrict__ WhT_f, const f16* __restrict__ WhT_b,
                  const float* __restrict__ bh_f, const float* __restrict__ bh_b,
                  f16* __restrict__ gruscat) {
    __shared__ f16   h16[B_][H_];
    __shared__ float h32[B_][H_];
    f16*   h16f = &h16[0][0];
    float* h32f = &h32[0][0];

    const int dir = blockIdx.x;
    const float* xp  = dir ? xp_b  : xp_f;
    const f16*   WhT = dir ? WhT_b : WhT_f;
    const float* bh  = dir ? bh_b  : bh_f;
    const int outOff = dir ? H_ : 0;

    for (int i = threadIdx.x; i < B_ * H_; i += blockDim.x) {
        h16f[i] = (f16)0;
        h32f[i] = 0.0f;
    }
    __syncthreads();

    const int lane = threadIdx.x & 31;
    const int wave = threadIdx.x >> 5;                 // 0..15
    const int m0   = ((wave * 2) >> 4) * 16;           // 0 or 16
    const int ntA  = (wave * 2) & 15;                  // even N-tile index

    for (int s = 0; s < N_; ++s) {
        const int tt = dir ? (N_ - 1 - s) : s;
        v8f c[2][3];
        rec_gemm6(h16f, WhT, m0, ntA, lane, c);

        float hnew[2][8];
#pragma unroll
        for (int tr = 0; tr < 2; ++tr) {
            const int col = (ntA + tr) * 16 + (lane & 15);
#pragma unroll
            for (int v = 0; v < 8; ++v) {
                const int b = m0 + 8 * (lane >> 4) + v;
                const float* xpp = xp + (size_t)(b * N_ + tt) * G3_;
                const float r  = sigmoidf_(xpp[col]          + c[tr][0][v] + bh[col]);
                const float z  = sigmoidf_(xpp[H_ + col]     + c[tr][1][v] + bh[H_ + col]);
                const float nn = tanhf   (xpp[2 * H_ + col]  + r * (c[tr][2][v] + bh[2 * H_ + col]));
                hnew[tr][v] = (1.0f - z) * nn + z * h32[b][col];
            }
        }
        __syncthreads();
#pragma unroll
        for (int tr = 0; tr < 2; ++tr) {
            const int col = (ntA + tr) * 16 + (lane & 15);
#pragma unroll
            for (int v = 0; v < 8; ++v) {
                const int b = m0 + 8 * (lane >> 4) + v;
                const float hv = hnew[tr][v];
                h32[b][col] = hv;
                h16[b][col] = (f16)hv;
                gruscat[(size_t)(b * N_ + tt) * (2 * H_) + outOff + col] = (f16)hv;
            }
        }
        __syncthreads();
    }
}

// ---------------------------------------------------------------------------
// Bottom-up (child-sum) tree GRU.  Single block; idx = 63..0.
// The f16 mirror of the child-sum state (Cbuf16) is DMA'd into LDS by the
// Tensor Data Mover each step: 2D tile 256(x) x 32(rows), row stride 64*256.
// ---------------------------------------------------------------------------
__global__ __launch_bounds__(512)
void k_dt_scan(const float* __restrict__ xp, const f16* __restrict__ UhT,
               const int* __restrict__ heads, float* __restrict__ Cbuf,
               f16* __restrict__ Cbuf16, float* __restrict__ Hout) {
    __shared__ f16 h16[B_][H_];
    f16* h16f = &h16[0][0];
    const int lane = threadIdx.x & 31;
    const int wave = threadIdx.x >> 5;
    const int m0   = ((wave * 2) >> 4) * 16;
    const int ntA  = (wave * 2) & 15;

    const unsigned int lds_off = (unsigned int)(uintptr_t)(void*)h16f;

    for (int s = N_ - 1; s >= 0; --s) {
        if (wave == 0) {
            // Tensor DMA: Cbuf16[:, s]  ->  LDS h16 (32 rows x 256 f16)
            const unsigned long long ga =
                (unsigned long long)(uintptr_t)(Cbuf16 + (size_t)s * H_);
            tdm_g0_t g0;
            g0[0] = 1u;                                   // count = 1 descriptor
            g0[1] = lds_off;                              // LDS byte address
            g0[2] = (unsigned int)(ga & 0xFFFFFFFFull);   // global addr lo
            g0[3] = (unsigned int)((ga >> 32) & 0x1FFFFFFull) | (2u << 30); // hi | type=2
            tdm_g1_t g1;
            g1[0] = 0x10000;                              // data_size = 1 (2 bytes)
            g1[1] = (int)((unsigned)H_ << 16);            // tensor_dim0 = 256 (bits 79:48)
            g1[2] = (int)((unsigned)B_ << 16);            // tensor_dim1 = 32  (bits 111:80)
            g1[3] = (int)((unsigned)H_ << 16);            // tile_dim0  = 256  (bits 127:112)
            g1[4] = B_;                                   // tile_dim1  = 32
            g1[5] = N_ * H_;                              // tensor_dim0_stride = 16384
            g1[6] = 0;
            g1[7] = 0;
            tdm_g2_t g2 = {0, 0, 0, 0};
#if __clang_major__ >= 23
            tdm_g1_t z8 = {};
            __builtin_amdgcn_tensor_load_to_lds(g0, g1, g2, g2, z8, 0);
#else
            __builtin_amdgcn_tensor_load_to_lds(g0, g1, g2, g2, 0);
#endif
            __builtin_amdgcn_s_wait_tensorcnt(0);
        }
        __syncthreads();

        v8f c[2][3];
        rec_gemm6(h16f, UhT, m0, ntA, lane, c);

        float hnew[2][8];
#pragma unroll
        for (int tr = 0; tr < 2; ++tr) {
            const int col = (ntA + tr) * 16 + (lane & 15);
#pragma unroll
            for (int v = 0; v < 8; ++v) {
                const int b = m0 + 8 * (lane >> 4) + v;
                const float* xpp = xp + (size_t)(b * N_ + s) * G3_;
                const float r  = sigmoidf_(xpp[col]         + c[tr][0][v]);
                const float z  = sigmoidf_(xpp[H_ + col]    + c[tr][1][v]);
                const float nn = tanhf   (xpp[2 * H_ + col] + r * c[tr][2][v]);
                const float hp = Cbuf[(size_t)(b * N_ + s) * H_ + col];
                hnew[tr][v] = (1.0f - z) * nn + z * hp;
            }
        }
        __syncthreads();
#pragma unroll
        for (int tr = 0; tr < 2; ++tr) {
            const int col = (ntA + tr) * 16 + (lane & 15);
#pragma unroll
            for (int v = 0; v < 8; ++v) {
                const int b = m0 + 8 * (lane >> 4) + v;
                const float hv = hnew[tr][v];
                Hout[(size_t)(b * N_ + s) * H_ + col] = hv;
                const int par = heads[b * N_ + s];
                if (par >= 0) {
                    const size_t ci = (size_t)(b * N_ + par) * H_ + col;
                    const float nv = Cbuf[ci] + hv;
                    Cbuf[ci]   = nv;
                    Cbuf16[ci] = (f16)nv;
                }
            }
        }
        __threadfence();
        __syncthreads();
    }
}

// ---------------------------------------------------------------------------
// Top-down tree GRU.  Single block; idx = 0..63.  hprev = Hout[b, heads[b,idx]]
// (gather-indexed per batch row, staged by threads).
// ---------------------------------------------------------------------------
__global__ __launch_bounds__(512)
void k_td_scan(const float* __restrict__ xp, const f16* __restrict__ UhT,
               const int* __restrict__ heads, float* __restrict__ Hout) {
    __shared__ f16   h16[B_][H_];
    __shared__ float hp32[B_][H_];
    f16*   h16f  = &h16[0][0];
    float* hp32f = &hp32[0][0];
    const int lane = threadIdx.x & 31;
    const int wave = threadIdx.x >> 5;
    const int m0   = ((wave * 2) >> 4) * 16;
    const int ntA  = (wave * 2) & 15;

    for (int s = 0; s < N_; ++s) {
        for (int i = threadIdx.x; i < B_ * H_; i += blockDim.x) {
            const int b = i >> 8, k = i & (H_ - 1);
            const int par = heads[b * N_ + s];
            const float v = (par >= 0) ? Hout[(size_t)(b * N_ + par) * H_ + k] : 0.0f;
            h16f[i]  = (f16)v;
            hp32f[i] = v;
        }
        __syncthreads();

        v8f c[2][3];
        rec_gemm6(h16f, UhT, m0, ntA, lane, c);

        float hnew[2][8];
#pragma unroll
        for (int tr = 0; tr < 2; ++tr) {
            const int col = (ntA + tr) * 16 + (lane & 15);
#pragma unroll
            for (int v = 0; v < 8; ++v) {
                const int b = m0 + 8 * (lane >> 4) + v;
                const float* xpp = xp + (size_t)(b * N_ + s) * G3_;
                const float r  = sigmoidf_(xpp[col]         + c[tr][0][v]);
                const float z  = sigmoidf_(xpp[H_ + col]    + c[tr][1][v]);
                const float nn = tanhf   (xpp[2 * H_ + col] + r * c[tr][2][v]);
                hnew[tr][v] = (1.0f - z) * nn + z * hp32[b][col];
            }
        }
        __syncthreads();
#pragma unroll
        for (int tr = 0; tr < 2; ++tr) {
            const int col = (ntA + tr) * 16 + (lane & 15);
#pragma unroll
            for (int v = 0; v < 8; ++v) {
                const int b = m0 + 8 * (lane >> 4) + v;
                Hout[(size_t)(b * N_ + s) * H_ + col] = hnew[tr][v];
            }
        }
        __threadfence();
        __syncthreads();
    }
}

// ---------------------------------------------------------------------------
// Max-pool over N of concat(dt,td) then fc (512->100) then fc2 (100->5).
// ---------------------------------------------------------------------------
__global__ __launch_bounds__(512)
void k_final(const float* __restrict__ dtO, const float* __restrict__ tdO,
             const float* __restrict__ fcW, const float* __restrict__ fcB,
             const float* __restrict__ fc2W, const float* __restrict__ fc2B,
             float* __restrict__ out) {
    __shared__ float ctx[2 * H_];
    __shared__ float y[100];
    const int b = blockIdx.x;
    const int tid = threadIdx.x;

    const float* src = (tid < H_) ? dtO : tdO;
    const int col = tid & (H_ - 1);
    float mx = -3.4e38f;
    for (int n = 0; n < N_; ++n)
        mx = fmaxf(mx, src[(size_t)(b * N_ + n) * H_ + col]);
    ctx[tid] = mx;
    __syncthreads();

    if (tid < 100) {
        float acc = fcB[tid];
        for (int j = 0; j < 2 * H_; ++j)
            acc += ctx[j] * fcW[(size_t)j * 100 + tid];
        y[tid] = acc;
    }
    __syncthreads();

    if (tid < 5) {
        float acc = fc2B[tid];
        for (int o = 0; o < 100; ++o)
            acc += y[o] * fc2W[o * 5 + tid];
        out[b * 5 + tid] = acc;
    }
}

// ---------------------------------------------------------------------------
extern "C" void kernel_launch(void* const* d_in, const int* in_sizes, int n_in,
                              void* d_out, int out_size, void* d_ws, size_t ws_size,
                              hipStream_t stream) {
    (void)in_sizes; (void)n_in; (void)out_size; (void)ws_size;

    const int*   xs     = (const int*)  d_in[0];
    const int*   heads  = (const int*)  d_in[1];
    const int*   rels   = (const int*)  d_in[2];
    const int*   labels = (const int*)  d_in[3];
    const float* emb_W  = (const float*)d_in[9];
    const float* rel_W  = (const float*)d_in[10];
    const float* Wi_f   = (const float*)d_in[11];
    const float* Wh_f   = (const float*)d_in[12];
    const float* bi_f   = (const float*)d_in[13];
    const float* bh_f   = (const float*)d_in[14];
    const float* Wi_b   = (const float*)d_in[15];
    const float* Wh_b   = (const float*)d_in[16];
    const float* bi_b   = (const float*)d_in[17];
    const float* bh_b   = (const float*)d_in[18];
    const float* Wt     = (const float*)d_in[19];
    const float* bt     = (const float*)d_in[20];
    const float* dt_Wx  = (const float*)d_in[21];
    const float* dt_Uh  = (const float*)d_in[22];
    const float* dt_b   = (const float*)d_in[23];
    const float* td_Wx  = (const float*)d_in[24];
    const float* td_Uh  = (const float*)d_in[25];
    const float* td_b   = (const float*)d_in[26];
    const float* fc_W   = (const float*)d_in[27];
    const float* fc_b   = (const float*)d_in[28];
    const float* fc2_W  = (const float*)d_in[29];
    const float* fc2_b  = (const float*)d_in[30];
    float* out = (float*)d_out;

    char* ws = (char*)d_ws;
    size_t off = 0;
    auto take = [&](size_t bytes) -> char* {
        char* p = ws + off;
        off = (off + bytes + 255) & ~(size_t)255;
        return p;
    };

    f16*   embH   = (f16*)  take((size_t)BN_ * 2 * D_ * 2);
    f16*   WiTf   = (f16*)  take((size_t)G3_ * 2 * D_ * 2);
    f16*   WiTb   = (f16*)  take((size_t)G3_ * 2 * D_ * 2);
    f16*   WhTf   = (f16*)  take((size_t)G3_ * H_ * 2);
    f16*   WhTb   = (f16*)  take((size_t)G3_ * H_ * 2);
    f16*   WtT    = (f16*)  take((size_t)D_ * 2 * H_ * 2);
    f16*   dtWxT  = (f16*)  take((size_t)G3_ * 2 * D_ * 2);
    f16*   tdWxT  = (f16*)  take((size_t)G3_ * 2 * D_ * 2);
    f16*   dtUhT  = (f16*)  take((size_t)G3_ * H_ * 2);
    f16*   tdUhT  = (f16*)  take((size_t)G3_ * H_ * 2);
    float* xp_f   = (float*)take((size_t)BN_ * G3_ * 4);
    float* xp_b   = (float*)take((size_t)BN_ * G3_ * 4);
    f16*   grusH  = (f16*)  take((size_t)BN_ * 2 * H_ * 2);
    float* grus   = (float*)take((size_t)BN_ * D_ * 4);
    f16*   outH   = (f16*)  take((size_t)BN_ * 2 * D_ * 2);
    float* dt_xp  = (float*)take((size_t)BN_ * G3_ * 4);
    float* td_xp  = (float*)take((size_t)BN_ * G3_ * 4);
    float* Cbuf   = (float*)take((size_t)BN_ * H_ * 4);
    f16*   Cbuf16 = (f16*)  take((size_t)BN_ * H_ * 2);
    float* dtO    = (float*)take((size_t)BN_ * H_ * 4);
    float* tdO    = (float*)take((size_t)BN_ * H_ * 4);

    const dim3 b256(256);
    auto gr1 = [](int total) { return dim3((total + 255) / 256); };

    // 1) weight transpose+convert to f16
    k_wtrans<<<gr1(2 * D_ * G3_), b256, 0, stream>>>(Wi_f,  WiTf,  2 * D_, G3_);
    k_wtrans<<<gr1(2 * D_ * G3_), b256, 0, stream>>>(Wi_b,  WiTb,  2 * D_, G3_);
    k_wtrans<<<gr1(H_ * G3_),     b256, 0, stream>>>(Wh_f,  WhTf,  H_,     G3_);
    k_wtrans<<<gr1(H_ * G3_),     b256, 0, stream>>>(Wh_b,  WhTb,  H_,     G3_);
    k_wtrans<<<gr1(2 * H_ * D_),  b256, 0, stream>>>(Wt,    WtT,   2 * H_, D_);
    k_wtrans<<<gr1(2 * D_ * G3_), b256, 0, stream>>>(dt_Wx, dtWxT, 2 * D_, G3_);
    k_wtrans<<<gr1(2 * D_ * G3_), b256, 0, stream>>>(td_Wx, tdWxT, 2 * D_, G3_);
    k_wtrans<<<gr1(H_ * G3_),     b256, 0, stream>>>(dt_Uh, dtUhT, H_,     G3_);
    k_wtrans<<<gr1(H_ * G3_),     b256, 0, stream>>>(td_Uh, tdUhT, H_,     G3_);

    // 2) embedding gather (word + label concat) -> f16
    k_gather<<<gr1(BN_ * 2 * D_), b256, 0, stream>>>(xs, labels, emb_W, embH);

    // 3) xp_f / xp_b = emb @ Wi + bi    (2048 x 768, K=512)
    k_gemm_bias<<<dim3(G3_ / 256, BN_ / 32), b256, 0, stream>>>(embH, WiTf, bi_f, xp_f, BN_, G3_, 2 * D_);
    k_gemm_bias<<<dim3(G3_ / 256, BN_ / 32), b256, 0, stream>>>(embH, WiTb, bi_b, xp_b, BN_, G3_, 2 * D_);

    // zero the child-sum accumulators for the bottom-up tree
    hipMemsetAsync(Cbuf,   0, (size_t)BN_ * H_ * 4, stream);
    hipMemsetAsync(Cbuf16, 0, (size_t)BN_ * H_ * 2, stream);

    // 4) bidirectional GRU scan (persistent, grid=2)
    k_bigru_scan<<<dim3(2), dim3(512), 0, stream>>>(xp_f, xp_b, WhTf, WhTb, bh_f, bh_b, grusH);

    // 5) grus = concat(fwd,bwd) @ Wt + bt   (2048 x 256, K=512)
    k_gemm_bias<<<dim3(D_ / 256, BN_ / 32), b256, 0, stream>>>(grusH, WtT, bt, grus, BN_, D_, 2 * H_);

    // 6) outputs = concat(grus, rel_emb) -> f16
    k_build_outputs<<<gr1(BN_ * 2 * D_), b256, 0, stream>>>(grus, rels, rel_W, outH);

    // 7) tree input projections (2048 x 768, K=512)
    k_gemm_bias<<<dim3(G3_ / 256, BN_ / 32), b256, 0, stream>>>(outH, dtWxT, dt_b, dt_xp, BN_, G3_, 2 * D_);
    k_gemm_bias<<<dim3(G3_ / 256, BN_ / 32), b256, 0, stream>>>(outH, tdWxT, td_b, td_xp, BN_, G3_, 2 * D_);

    // 8) tree scans (persistent single block each; dt uses the Tensor Data Mover)
    k_dt_scan<<<dim3(1), dim3(512), 0, stream>>>(dt_xp, dtUhT, heads, Cbuf, Cbuf16, dtO);
    k_td_scan<<<dim3(1), dim3(512), 0, stream>>>(td_xp, tdUhT, heads, tdO);

    // 9) max-pool + fc + fc2
    k_final<<<dim3(B_), dim3(512), 0, stream>>>(dtO, tdO, fc_W, fc_b, fc2_W, fc2_b, out);
}
